// LengthRegulator_45578192945368
// MI455X (gfx1250) — matrix operand order repaired and checked by reading, not compiled, not run on hardware
//
#include <hip/hip_runtime.h>
#include <stdint.h>

// Problem constants (from reference)
#define B_    16
#define L_    1024
#define D_    384
#define F_    1536
#define MAXDUR 10
#define TOUT  (L_ * MAXDUR)
#define EPSV  1e-5f

typedef __attribute__((ext_vector_type(16))) __bf16 v16bf;
typedef __attribute__((ext_vector_type(8)))  float  v8f;

__device__ __forceinline__ unsigned short f2bf(float f) {
  union { float f; unsigned int u; } c; c.f = f;
  unsigned int u = c.u;
  unsigned int r = (u + 0x7FFFu + ((u >> 16) & 1u)) >> 16;   // RNE
  return (unsigned short)r;
}
__device__ __forceinline__ float bf2f(unsigned short h) {
  union { unsigned int u; float f; } c; c.u = ((unsigned int)h) << 16;
  return c.f;
}
// Generic pointers to LDS carry the group-relative byte offset in addr[31:0]
__device__ __forceinline__ unsigned lds_off(const void* p) {
  return (unsigned)(unsigned long long)p;
}
__device__ __forceinline__ void async_copy_b128(unsigned lds_addr, const void* gaddr) {
  asm volatile("global_load_async_to_lds_b128 %0, %1, off"
               :: "v"(lds_addr), "v"((unsigned long long)gaddr) : "memory");
}
__device__ __forceinline__ void wait_async0() {
  asm volatile("s_wait_asynccnt 0x0" ::: "memory");
}

// ---------------------------------------------------------------------------
// fp32 -> bf16 flat convert (for y)
// ---------------------------------------------------------------------------
__global__ void cvt_y_kernel(const float* __restrict__ in,
                             unsigned short* __restrict__ out, int n) {
  int i = blockIdx.x * blockDim.x + threadIdx.x;
  if (i < n) out[i] = f2bf(in[i]);
}

// ---------------------------------------------------------------------------
// Weight reorg: (O, I, 3) fp32 -> [k][o][i] bf16
// ---------------------------------------------------------------------------
__global__ void cvt_w_kernel(const float* __restrict__ in,
                             unsigned short* __restrict__ out, int O, int I) {
  int idx = blockIdx.x * blockDim.x + threadIdx.x;
  int n = 3 * O * I;
  if (idx >= n) return;
  int k = idx / (O * I);
  int rem = idx - k * O * I;
  int o = rem / I;
  int i = rem - o * I;
  out[idx] = f2bf(in[(size_t)o * I * 3 + (size_t)i * 3 + k]);
}

// ---------------------------------------------------------------------------
// WMMA conv1d (K=3, SAME, bias+ReLU), implicit GEMM decomposed over taps.
//   in  : (B, L, CI) bf16
//   wts : (3, CO, CI) bf16    (tap-major, Cout rows, Cin contiguous)
//   out : (B, L, CO) bf16
// Workgroup: 256 thr = 8 waves; tile 64(M=pos) x 128(N=outch).
// Wave (wm, wn): M rows [16*wm,+16), N cols [64*wn,+64) -> 4 accumulators.
// Double-buffered LDS, async global->LDS DMA, one barrier per 32-wide K step.
// CI/CO compile-time so ksteps, strides and trip counts are constants.
// ---------------------------------------------------------------------------
template <int CI, int CO>
__global__ __launch_bounds__(256) void conv_wmma_kernel(
    const unsigned short* __restrict__ in,
    const unsigned short* __restrict__ wts,
    const float* __restrict__ bias,
    unsigned short* __restrict__ out) {
  // Row stride 40 halves = 80 B (16B-aligned, decorrelates LDS banks)
  __shared__ unsigned short Ald[2][64][40];
  __shared__ unsigned short Bld[2][128][40];

  constexpr int KSTEPS = CI >> 5;      // 32-wide K steps per tap
  constexpr int S      = 3 * KSTEPS;   // total pipeline steps

  const int b    = blockIdx.z;
  const int m0   = blockIdx.x * 64;
  const int n0   = blockIdx.y * 128;
  const int tid  = threadIdx.x;
  const int lane = tid & 31;
  const int wave = tid >> 5;
  const int wm   = wave & 3;    // M sub-tile (16 rows)
  const int wn   = wave >> 2;   // N half (64 cols)
  const int half = lane >> 4;   // K-half selector per ISA layout
  const int l15  = lane & 15;

  const int ar = tid >> 2;      // staged tile row 0..63
  const int ag = tid & 3;       // 8-halfword group in row

  // Uniform flag: only the first/last M tile can touch the SAME-padding halo
  const bool edge = (m0 == 0) || (m0 + 64 >= L_);

  v8f acc[4] = {v8f{}, v8f{}, v8f{}, v8f{}};

  auto issue = [&](int k, int kk, int p) {
    const int coff = (kk << 5) + (ag << 3);
    // B tile: 128 rows (out channels) x 32 halves; always in bounds
    const unsigned short* gb0 =
        wts + (size_t)(k * CO + n0 + ar) * (size_t)CI + coff;
    async_copy_b128(lds_off(&Bld[p][ar][ag << 3]), gb0);
    async_copy_b128(lds_off(&Bld[p][ar + 64][ag << 3]), gb0 + (size_t)64 * CI);
    // A tile: 64 rows x 32 halves; zero-fill halo rows on edge tiles only
    const int lsrc = m0 + ar + k - 1;
    const unsigned short* ga =
        in + (size_t)(b * L_ + lsrc) * (size_t)CI + coff;
    if (!edge) {
      async_copy_b128(lds_off(&Ald[p][ar][ag << 3]), ga);
    } else if (lsrc >= 0 && lsrc < L_) {
      async_copy_b128(lds_off(&Ald[p][ar][ag << 3]), ga);
    } else {
      *reinterpret_cast<uint4*>(&Ald[p][ar][ag << 3]) = make_uint4(0u, 0u, 0u, 0u);
    }
  };

  int nk = 0, nkk = 0;
  issue(nk, nkk, 0);
  if (++nkk == KSTEPS) { nkk = 0; ++nk; }

#pragma unroll 2
  for (int s = 0; s < S; ++s) {
    wait_async0();            // my buf[s&1] transfers landed
    __syncthreads();          // everyone's landed; everyone done reading buf[(s+1)&1]
    if (s + 1 < S) {
      issue(nk, nkk, (s + 1) & 1);   // DMA next tile, overlapped with compute
      if (++nkk == KSTEPS) { nkk = 0; ++nk; }
    }
    const int p = s & 1;
    // A fragment (16x32): lanes 0-15 -> K{0..7,16..23}; lanes 16-31 -> K{8..15,24..31}
    union { uint4 q[2]; v16bf v; } af, bfr;
    const int arow = wm * 16 + l15;
    af.q[0] = *reinterpret_cast<const uint4*>(&Ald[p][arow][half << 3]);
    af.q[1] = *reinterpret_cast<const uint4*>(&Ald[p][arow][16 + (half << 3)]);
#pragma unroll
    for (int j = 0; j < 4; ++j) {
      // B fragment (32x16): lane = column; lanes 0-15 -> K{0..7,16..23}; lanes 16-31 -> K{8..15,24..31}
      const int brow = wn * 64 + j * 16 + l15;
      bfr.q[0] = *reinterpret_cast<const uint4*>(&Bld[p][brow][half << 3]);
      bfr.q[1] = *reinterpret_cast<const uint4*>(&Bld[p][brow][16 + (half << 3)]);
      acc[j] = __builtin_amdgcn_wmma_f32_16x16x32_bf16(
          false, af.v, false, bfr.v, (short)0, acc[j], false, false);
    }
  }

  // Epilogue: C/D layout — VGPR r holds M=r (lanes 0-15) / M=r+8 (lanes 16-31), N = lane%16
#pragma unroll
  for (int j = 0; j < 4; ++j) {
    const int ng = n0 + wn * 64 + j * 16 + l15;
    const float bv = bias[ng];
#pragma unroll
    for (int r = 0; r < 8; ++r) {
      const int m = wm * 16 + r + half * 8;
      float v = acc[j][r] + bv;
      v = v > 0.f ? v : 0.f;
      out[(size_t)(b * L_ + m0 + m) * (size_t)CO + ng] = f2bf(v);
    }
  }
}

// ---------------------------------------------------------------------------
// LayerNorm over D=384 (bf16 in -> bf16 out), one row per block (128 thr)
// ---------------------------------------------------------------------------
__global__ __launch_bounds__(128) void ln_kernel(
    const unsigned short* __restrict__ in,
    const float* __restrict__ g, const float* __restrict__ be,
    unsigned short* __restrict__ out) {
  __shared__ float s1[128], s2[128];
  const size_t row = (size_t)blockIdx.x * D_;
  const int tid = threadIdx.x;
  float x0 = bf2f(in[row + tid]);
  float x1 = bf2f(in[row + tid + 128]);
  float x2 = bf2f(in[row + tid + 256]);
  s1[tid] = x0 + x1 + x2;
  s2[tid] = x0 * x0 + x1 * x1 + x2 * x2;
  __syncthreads();
  for (int off = 64; off > 0; off >>= 1) {
    if (tid < off) { s1[tid] += s1[tid + off]; s2[tid] += s2[tid + off]; }
    __syncthreads();
  }
  const float mu  = s1[0] * (1.0f / D_);
  const float var = s2[0] * (1.0f / D_) - mu * mu;
  const float rs  = rsqrtf(var + EPSV);
  out[row + tid]       = f2bf((x0 - mu) * rs * g[tid]       + be[tid]);
  out[row + tid + 128] = f2bf((x1 - mu) * rs * g[tid + 128] + be[tid + 128]);
  out[row + tid + 256] = f2bf((x2 - mu) * rs * g[tid + 256] + be[tid + 256]);
}

// ---------------------------------------------------------------------------
// Duration: p = h.wl + bl; d = clip(round(exp(p)), 0, 10); mask l>=tok -> 0
// ---------------------------------------------------------------------------
__global__ __launch_bounds__(128) void dur_kernel(
    const unsigned short* __restrict__ h,
    const float* __restrict__ wl, const float* __restrict__ bl,
    const int* __restrict__ tok, int* __restrict__ lns) {
  __shared__ float s[128];
  const int rowi = blockIdx.x;
  const int b = rowi / L_;
  const int l = rowi - b * L_;
  const size_t row = (size_t)rowi * D_;
  const int tid = threadIdx.x;
  float p = bf2f(h[row + tid])       * wl[tid]
          + bf2f(h[row + tid + 128]) * wl[tid + 128]
          + bf2f(h[row + tid + 256]) * wl[tid + 256];
  s[tid] = p;
  __syncthreads();
  for (int off = 64; off > 0; off >>= 1) {
    if (tid < off) s[tid] += s[tid + off];
    __syncthreads();
  }
  if (tid == 0) {
    float e = expf(s[0] + bl[0]);
    float r = roundf(e);
    if (r < 0.f) r = 0.f;
    if (r > (float)MAXDUR) r = (float)MAXDUR;
    int d = (int)r;
    if (l >= tok[b]) d = 0;
    lns[rowi] = d;
  }
}

// ---------------------------------------------------------------------------
// Inclusive scan over L=1024 per batch (Hillis-Steele in LDS) + totals
// ---------------------------------------------------------------------------
__global__ __launch_bounds__(1024) void scan_kernel(
    const int* __restrict__ lns, int* __restrict__ csum,
    int* __restrict__ totals, float* __restrict__ out_totals) {
  __shared__ int s[1024];
  const int b = blockIdx.x;
  const int tid = threadIdx.x;
  s[tid] = lns[b * L_ + tid];
  __syncthreads();
  for (int off = 1; off < 1024; off <<= 1) {
    int add = (tid >= off) ? s[tid - off] : 0;
    __syncthreads();
    s[tid] += add;
    __syncthreads();
  }
  csum[b * L_ + tid] = s[tid];
  if (tid == 0) {
    totals[b] = s[1023];
    out_totals[b] = (float)s[1023];
  }
}

// ---------------------------------------------------------------------------
// Length-regulated gather: one block per (t, b), 96 thr x float4 = 384 floats
// ---------------------------------------------------------------------------
__global__ __launch_bounds__(96) void gather_kernel(
    const float* __restrict__ y, const int* __restrict__ csum,
    const int* __restrict__ totals, float* __restrict__ out) {
  const int t = blockIdx.x;
  const int b = blockIdx.y;
  const int* c = csum + b * L_;
  int lo = 0, hi = L_;
  while (lo < hi) {                       // searchsorted side='right'
    int mid = (lo + hi) >> 1;
    if (c[mid] <= t) lo = mid + 1; else hi = mid;
  }
  int idx = lo < L_ ? lo : (L_ - 1);
  const bool valid = t < totals[b];
  float4 v = make_float4(0.f, 0.f, 0.f, 0.f);
  if (valid)
    v = reinterpret_cast<const float4*>(y + (size_t)(b * L_ + idx) * D_)[threadIdx.x];
  reinterpret_cast<float4*>(out + (size_t)(b * TOUT + t) * D_)[threadIdx.x] = v;
}

// ---------------------------------------------------------------------------
extern "C" void kernel_launch(void* const* d_in, const int* in_sizes, int n_in,
                              void* d_out, int out_size, void* d_ws, size_t ws_size,
                              hipStream_t stream) {
  const float* y   = (const float*)d_in[0];
  const int*   tok = (const int*)  d_in[1];
  const float* w1a = (const float*)d_in[2];  const float* b1a = (const float*)d_in[3];
  const float* w1b = (const float*)d_in[4];  const float* b1b = (const float*)d_in[5];
  const float* g1  = (const float*)d_in[6];  const float* be1 = (const float*)d_in[7];
  const float* w2a = (const float*)d_in[8];  const float* b2a = (const float*)d_in[9];
  const float* w2b = (const float*)d_in[10]; const float* b2b = (const float*)d_in[11];
  const float* g2  = (const float*)d_in[12]; const float* be2 = (const float*)d_in[13];
  const float* wl  = (const float*)d_in[14]; const float* bl  = (const float*)d_in[15];
  float* out = (float*)d_out;

  char* ws = (char*)d_ws;
  size_t off = 0;
  auto alloc = [&](size_t bytes) -> char* {
    char* p = ws + off;
    off += (bytes + 255) & ~(size_t)255;
    return p;
  };
  const size_t WELEMS = (size_t)3 * F_ * D_;                 // per conv weight
  unsigned short* wb1a = (unsigned short*)alloc(WELEMS * 2);
  unsigned short* wb1b = (unsigned short*)alloc(WELEMS * 2);
  unsigned short* wb2a = (unsigned short*)alloc(WELEMS * 2);
  unsigned short* wb2b = (unsigned short*)alloc(WELEMS * 2);
  unsigned short* ybf  = (unsigned short*)alloc((size_t)B_ * L_ * D_ * 2);
  unsigned short* actF = (unsigned short*)alloc((size_t)B_ * L_ * F_ * 2);
  unsigned short* actD = (unsigned short*)alloc((size_t)B_ * L_ * D_ * 2);
  unsigned short* actDn= (unsigned short*)alloc((size_t)B_ * L_ * D_ * 2);
  int* lns    = (int*)alloc((size_t)B_ * L_ * 4);
  int* csum   = (int*)alloc((size_t)B_ * L_ * 4);
  int* totals = (int*)alloc((size_t)B_ * 4);
  (void)ws_size; (void)in_sizes; (void)n_in; (void)out_size;

  // 1) bf16 conversions / weight reorg
  {
    int n = B_ * L_ * D_;
    cvt_y_kernel<<<(n + 255) / 256, 256, 0, stream>>>(y, ybf, n);
    int nw = (int)WELEMS;
    cvt_w_kernel<<<(nw + 255) / 256, 256, 0, stream>>>(w1a, wb1a, F_, D_);
    cvt_w_kernel<<<(nw + 255) / 256, 256, 0, stream>>>(w1b, wb1b, D_, F_);
    cvt_w_kernel<<<(nw + 255) / 256, 256, 0, stream>>>(w2a, wb2a, F_, D_);
    cvt_w_kernel<<<(nw + 255) / 256, 256, 0, stream>>>(w2b, wb2b, D_, F_);
  }

  // 2) conv blocks via bf16 WMMA (tiles: 64 pos x 128 outch)
  dim3 gA(L_ / 64, F_ / 128, B_);   // D -> F
  dim3 gB(L_ / 64, D_ / 128, B_);   // F -> D
  conv_wmma_kernel<D_, F_><<<gA, 256, 0, stream>>>(ybf,  wb1a, b1a, actF);
  conv_wmma_kernel<F_, D_><<<gB, 256, 0, stream>>>(actF, wb1b, b1b, actD);
  ln_kernel<<<B_ * L_, 128, 0, stream>>>(actD, g1, be1, actDn);
  conv_wmma_kernel<D_, F_><<<gA, 256, 0, stream>>>(actDn, wb2a, b2a, actF);
  conv_wmma_kernel<F_, D_><<<gB, 256, 0, stream>>>(actF,  wb2b, b2b, actD);
  ln_kernel<<<B_ * L_, 128, 0, stream>>>(actD, g2, be2, actDn);

  // 3) durations, scan, gather
  dur_kernel<<<B_ * L_, 128, 0, stream>>>(actDn, wl, bl, tok, lns);
  scan_kernel<<<B_, 1024, 0, stream>>>(lns, csum, totals,
                                       out + (size_t)B_ * TOUT * D_);
  gather_kernel<<<dim3(TOUT, B_), 96, 0, stream>>>(y, csum, totals, out);
}
